// VI2DSSM_14714557956427
// MI455X (gfx1250) — matrix operand order repaired and checked by reading, not compiled, not run on hardware
//
#include <hip/hip_runtime.h>

#define BSZ  4
#define DIN  256
#define SEQL 2048
#define NST  16
#define RR   16
#define PKK  (RR + 2 * NST)   // 48 projection rows
#define TL   64               // l-tile per GEMM wave (4 x 16 subtiles)

typedef __attribute__((ext_vector_type(16))) _Float16 v16h;
typedef __attribute__((ext_vector_type(4)))  _Float16 v4h;
typedef __attribute__((ext_vector_type(8)))  float    v8f;

// ---------------------------------------------------------------------------
// xmean[b,l] = mean over d of x[b,d,l]
// ---------------------------------------------------------------------------
__global__ void mean_kernel(const float* __restrict__ x, float* __restrict__ xmean) {
    int idx = blockIdx.x * blockDim.x + threadIdx.x;   // over B*L
    if (idx >= BSZ * SEQL) return;
    int b = idx / SEQL, l = idx % SEQL;
    const float* p = x + (long)b * DIN * SEQL + l;
    float s = 0.f;
    for (int d = 0; d < DIN; ++d) s += p[(long)d * SEQL];
    xmean[idx] = s * (1.0f / DIN);
}

// ---------------------------------------------------------------------------
// psi_coef[n] = sum_d Psi_w[n,d] * W_v[d]   (psi_B[b,n,l] = xmean[b,l]*coef[n])
// ---------------------------------------------------------------------------
__global__ void psi_coef_kernel(const float* __restrict__ Psi_hh,
                                const float* __restrict__ Psi_hv,
                                const float* __restrict__ Wv,
                                float* __restrict__ coef_hh,
                                float* __restrict__ coef_hv) {
    int t = threadIdx.x;
    if (t < NST) {
        float s = 0.f;
        for (int d = 0; d < DIN; ++d) s += Psi_hh[t * DIN + d] * Wv[d];
        coef_hh[t] = s;
    } else if (t < 2 * NST) {
        int n = t - NST;
        float s = 0.f;
        for (int d = 0; d < DIN; ++d) s += Psi_hv[n * DIN + d] * Wv[d];
        coef_hv[n] = s;
    }
}

// ---------------------------------------------------------------------------
// WMMA GEMM:  OUT[b, m, l] (+)= sum_k W[m*ldw + k] * X[b*x_bstride + k*L + l]
// One wave computes a 16(m) x 64(l) tile: 1 A fragment shared by 4 B
// fragments -> 4 v_wmma_f32_16x16x32_f16 per 32-wide K step.
// Tiles are staged into LDS in FRAGMENT ORDER (each lane's 16 halves
// contiguous) so fragment loads are ds_load_b128 pairs.
//   A elem (m,k): lane = (k>=16)*16 + m,    e = k & 15
//   B elem (k,l): lane = (k>=16)*16 + l,    e = k & 15
// Staging is float4-vectorized; K edge zero-pad is branchless (clamped
// address + vector select). Requires M % 16 == 0 and Kdim % 16 == 0
// (true for all call sites: M in {16,48,256}, Kdim in {16,256}).
// ---------------------------------------------------------------------------
__global__ void wmma_gemm_wxl(const float* __restrict__ W, int M, int Kdim, int ldw,
                              const float* __restrict__ X, long x_bstride,
                              float* __restrict__ OUT, long o_bstride,
                              int accumulate) {
    __shared__ __align__(32) _Float16 aF[32 * 16];      // A fragment slots
    __shared__ __align__(32) _Float16 bF[4][32 * 16];   // 4 B subtiles, fragment order

    const int lane = threadIdx.x;          // 0..31, one wave
    const int half = lane >> 4;
    const int lm   = lane & 15;
    const int m0 = blockIdx.x * 16;
    const int l0 = blockIdx.y * TL;
    const int b  = blockIdx.z;
    const float* Xb = X + (long)b * x_bstride;

    v8f acc0 = {}, acc1 = {}, acc2 = {}, acc3 = {};
    const v4h h4z = {};

    for (int k0 = 0; k0 < Kdim; k0 += 32) {
        // Prefetch next X tile rows into L2 (global_prefetch_b8)
        if (k0 + 32 < Kdim)
            __builtin_prefetch(&Xb[(long)(k0 + 32) * SEQL + l0], 0, 1);

        // ---- stage A (16 m x 32 k): float4 load -> v4h -> ds_store_b64 ----
        for (int vi = lane; vi < 128; vi += 32) {
            int r  = vi >> 3;                 // m offset 0..15
            int cg = (vi & 7) * 4;            // k offset group: 0,4,...,28
            int kk = k0 + cg;
            int kc = (kk < Kdim) ? kk : 0;    // clamped, always valid (Kdim%16==0)
            float4 w4 = *(const float4*)&W[(long)(m0 + r) * ldw + kc];
            v4h h4;
            h4[0] = (_Float16)w4.x; h4[1] = (_Float16)w4.y;
            h4[2] = (_Float16)w4.z; h4[3] = (_Float16)w4.w;
            h4 = (kk < Kdim) ? h4 : h4z;
            int slane = (cg >> 4) * 16 + r;
            int eb    = cg & 15;   // 4-aligned
            *(v4h*)(aF + slane * 16 + eb) = h4;
        }
        // ---- stage B (32 k x 64 l): float4 load along l, b16 scatter ----
        for (int vi = lane; vi < 512; vi += 32) {
            int r  = vi >> 4;                 // k offset 0..31
            int cg = (vi & 15) * 4;           // l offset group: 0,4,...,60
            int kk = k0 + r;
            int kc = (kk < Kdim) ? kk : 0;
            float4 x4 = *(const float4*)&Xb[(long)kc * SEQL + (l0 + cg)];
            float vals[4] = {x4.x, x4.y, x4.z, x4.w};
            int sub    = cg >> 4;             // constant within the 4-group
            int slbase = (r >> 4) << 4;
            #pragma unroll
            for (int j = 0; j < 4; ++j) {
                _Float16 hv = (kk < Kdim) ? (_Float16)vals[j] : (_Float16)0.f;
                bF[sub][(slbase + (cg & 15) + j) * 16 + (r & 15)] = hv;
            }
        }
        __syncthreads();

        // ---- fragment loads: contiguous 32B per lane -> ds_load_b128 x2 ----
        v16h a  = *(const v16h*)(aF    + lane * 16);
        v16h b0 = *(const v16h*)(bF[0] + lane * 16);
        v16h b1 = *(const v16h*)(bF[1] + lane * 16);
        v16h b2 = *(const v16h*)(bF[2] + lane * 16);
        v16h b3 = *(const v16h*)(bF[3] + lane * 16);

        acc0 = __builtin_amdgcn_wmma_f32_16x16x32_f16(false, a, false, b0,
                                                      (short)0, acc0, false, false);
        acc1 = __builtin_amdgcn_wmma_f32_16x16x32_f16(false, a, false, b1,
                                                      (short)0, acc1, false, false);
        acc2 = __builtin_amdgcn_wmma_f32_16x16x32_f16(false, a, false, b2,
                                                      (short)0, acc2, false, false);
        acc3 = __builtin_amdgcn_wmma_f32_16x16x32_f16(false, a, false, b3,
                                                      (short)0, acc3, false, false);
        __syncthreads();
    }

    // ---- write 16 x 64 tile: C/D vgpr j -> m = m0 + 8*half + j, l-col = lane&15
    float* Ob = OUT + (long)b * o_bstride;
    #pragma unroll
    for (int s = 0; s < 4; ++s) {
        v8f acc = (s == 0) ? acc0 : (s == 1) ? acc1 : (s == 2) ? acc2 : acc3;
        #pragma unroll
        for (int j = 0; j < 8; ++j) {
            int m = m0 + half * 8 + j;
            long idx = (long)m * SEQL + (l0 + s * 16 + lm);
            if (accumulate) Ob[idx] += acc[j];
            else            Ob[idx]  = acc[j];
        }
    }
}

// ---------------------------------------------------------------------------
// Selective scan: one lane per (b, d, n) state; 16-lane shuffle reduction
// over n produces y[b,d,l] = sum_n h*C + u*D. blockDim = 256 -> 16 (b,d) pairs.
// ---------------------------------------------------------------------------
__global__ void scan_kernel(const float* __restrict__ u,        // (B,DIN,L)
                            const float* __restrict__ dtraw,    // (B,DIN,L)
                            const float* __restrict__ dt_bias,  // (DIN)
                            const float* __restrict__ A_log,    // (DIN,N)
                            const float* __restrict__ Dp,       // (DIN)
                            const float* __restrict__ P,        // (B,PKK,L)
                            const float* __restrict__ xmean,    // (B,L)
                            const float* __restrict__ psi_coef, // (N)
                            const float* __restrict__ xB,       // (B,N,L) or null
                            float* __restrict__ out)            // (B,DIN,L)
{
    int pair = blockIdx.x * (blockDim.x >> 4) + (threadIdx.x >> 4);
    int n = threadIdx.x & 15;
    int b = pair / DIN, d = pair % DIN;

    float A    = -__expf(A_log[d * NST + n]);
    float bias = dt_bias[d];
    float Dd   = Dp[d];
    float pc   = psi_coef[n];

    const float* ub  = u     + ((long)b * DIN + d) * SEQL;
    const float* dtb = dtraw + ((long)b * DIN + d) * SEQL;
    const float* Bp  = P + ((long)b * PKK + RR + n) * SEQL;
    const float* Cp  = P + ((long)b * PKK + RR + NST + n) * SEQL;
    const float* xmb = xmean + (long)b * SEQL;
    const float* xBp = xB ? (xB + ((long)b * NST + n) * SEQL) : nullptr;
    float* ob = out + ((long)b * DIN + d) * SEQL;

    float h = 0.f;
    for (int l = 0; l < SEQL; ++l) {
        float uu = ub[l];
        float t  = dtb[l] + bias;
        float delta = (t > 20.f) ? t : log1pf(__expf(t));   // softplus
        float xbv = xBp ? xBp[l] : 0.f;
        float Bm = Bp[l] + xmb[l] * pc + xbv;
        float Cm = Cp[l] + xbv;
        float dA = __expf(delta * A);
        h = dA * h + delta * Bm * uu;
        float part = h * Cm;
        part += __shfl_xor(part, 8);
        part += __shfl_xor(part, 4);
        part += __shfl_xor(part, 2);
        part += __shfl_xor(part, 1);
        if (n == 0) ob[l] = part + uu * Dd;
    }
}

// ---------------------------------------------------------------------------
extern "C" void kernel_launch(void* const* d_in, const int* in_sizes, int n_in,
                              void* d_out, int out_size, void* d_ws, size_t ws_size,
                              hipStream_t stream) {
    const float* x        = (const float*)d_in[0];
    const float* xp_hh_w  = (const float*)d_in[1];   // (48,256)
    const float* dtp_hh_w = (const float*)d_in[2];   // (256,16)
    const float* dtp_hh_b = (const float*)d_in[3];   // (256)
    const float* Alog_hh  = (const float*)d_in[4];   // (256,16)
    const float* D_hh     = (const float*)d_in[5];   // (256)
    const float* xp_hv_w  = (const float*)d_in[6];
    const float* dtp_hv_w = (const float*)d_in[7];
    const float* dtp_hv_b = (const float*)d_in[8];
    const float* Alog_hv  = (const float*)d_in[9];
    const float* D_hv     = (const float*)d_in[10];
    const float* Wv       = (const float*)d_in[11];  // (256,1)
    const float* Psi_hh   = (const float*)d_in[12];  // (16,256)
    const float* Psi_hv   = (const float*)d_in[13];
    const float* X_hv     = (const float*)d_in[14];  // (16,256)
    const float* C_h      = (const float*)d_in[15];  // (256,256)
    const float* C_v      = (const float*)d_in[16];
    float* out = (float*)d_out;                      // (B,256,L)

    // Workspace carve-up (fp32 elements)
    float* ws = (float*)d_ws;
    size_t off = 0;
    auto alloc = [&](size_t n) { float* p = ws + off; off += n; return p; };
    float* xmean   = alloc((size_t)BSZ * SEQL);
    float* coef_hh = alloc(32);
    float* coef_hv = alloc(32);
    float* P1      = alloc((size_t)BSZ * PKK * SEQL);
    float* P2      = alloc((size_t)BSZ * PKK * SEQL);
    float* xBbuf   = alloc((size_t)BSZ * NST * SEQL);
    float* dt1     = alloc((size_t)BSZ * DIN * SEQL);
    float* dt2     = alloc((size_t)BSZ * DIN * SEQL);
    float* h_h     = alloc((size_t)BSZ * DIN * SEQL);
    float* h_v     = alloc((size_t)BSZ * DIN * SEQL);

    const long xBS  = (long)DIN * SEQL;   // x / h / dt batch stride
    const long pBS  = (long)PKK * SEQL;   // projection batch stride
    const long nBS  = (long)NST * SEQL;   // xB batch stride

    // ---- stage 0: small setup ----
    mean_kernel<<<(BSZ * SEQL + 255) / 256, 256, 0, stream>>>(x, xmean);
    psi_coef_kernel<<<1, 32, 0, stream>>>(Psi_hh, Psi_hv, Wv, coef_hh, coef_hv);

    dim3 gP (PKK / 16, SEQL / TL, BSZ);   // M=48
    dim3 gD (DIN / 16, SEQL / TL, BSZ);   // M=256
    dim3 gN (NST / 16, SEQL / TL, BSZ);   // M=16

    // ---- stage 1: horizontal pass ----
    wmma_gemm_wxl<<<gP, 32, 0, stream>>>(xp_hh_w, PKK, DIN, DIN, x, xBS, P1, pBS, 0);
    wmma_gemm_wxl<<<gD, 32, 0, stream>>>(dtp_hh_w, DIN, RR, RR, P1, pBS, dt1, xBS, 0);
    wmma_gemm_wxl<<<gN, 32, 0, stream>>>(X_hv, NST, DIN, DIN, x, xBS, xBbuf, nBS, 0);
    scan_kernel<<<BSZ * DIN / 16, 256, 0, stream>>>(
        x, dt1, dtp_hh_b, Alog_hh, D_hh, P1, xmean, coef_hh, nullptr, h_h);

    // ---- stage 2: vertical pass ----
    wmma_gemm_wxl<<<gP, 32, 0, stream>>>(xp_hv_w, PKK, DIN, DIN, h_h, xBS, P2, pBS, 0);
    wmma_gemm_wxl<<<gD, 32, 0, stream>>>(dtp_hv_w, DIN, RR, RR, P2, pBS, dt2, xBS, 0);
    scan_kernel<<<BSZ * DIN / 16, 256, 0, stream>>>(
        h_h, dt2, dtp_hv_b, Alog_hv, D_hv, P2, xmean, coef_hv, xBbuf, h_v);

    // ---- stage 3: output projection (dominant FLOPs, WMMA) ----
    wmma_gemm_wxl<<<gD, 32, 0, stream>>>(C_h, DIN, DIN, DIN, h_h, xBS, out, xBS, 0);
    wmma_gemm_wxl<<<gD, 32, 0, stream>>>(C_v, DIN, DIN, DIN, h_v, xBS, out, xBS, 1);
}
